// ACE_77876347011078
// MI455X (gfx1250) — matrix operand order repaired and checked by reading, not compiled, not run on hardware
//
#include <hip/hip_runtime.h>
#include <stdint.h>

typedef unsigned short u16;
typedef __bf16 bf16_t;
typedef bf16_t v16bf __attribute__((ext_vector_type(16)));
typedef float  v8f   __attribute__((ext_vector_type(8)));

#define EPSV 1e-5f
#define NCLS 35
#define SB 4
#define SC 256
#define SH 128
#define SW_ 256
#define SS 128   // style len / actv channels

static __device__ __forceinline__ u16 f2bf(float f) {
    unsigned int u = __float_as_uint(f);
    u = (u + 0x7FFFu + ((u >> 16) & 1u)) >> 16;
    return (u16)u;
}

// ---------------- K0: instance-norm stats: mean/rstd per (b,c) ----------------
__global__ void k_stats(const float* __restrict__ x, float* __restrict__ mean,
                        float* __restrict__ rstd) {
    __shared__ float ss[256], sq[256];
    int bc = blockIdx.x, t = threadIdx.x;
    const float* p = x + (size_t)bc * (SH * SW_);
    float s = 0.f, q = 0.f;
    for (int r = 0; r < 128; ++r) { float v = p[r * 256 + t]; s += v; q += v * v; }
    ss[t] = s; sq[t] = q; __syncthreads();
    for (int off = 128; off; off >>= 1) {
        if (t < off) { ss[t] += ss[t + off]; sq[t] += sq[t + off]; }
        __syncthreads();
    }
    if (t == 0) {
        float m = ss[0] * (1.f / 32768.f);
        float v = sq[0] * (1.f / 32768.f) - m * m;
        mean[bc] = m; rstd[bc] = rsqrtf(v + EPSV);
    }
}

// ---------------- K1: mu[b,j,o] = relu(fc_w[j]@style[b,j] + fc_b[j]) ----------
__global__ void k_mu(const float* __restrict__ style, const float* __restrict__ fcw,
                     const float* __restrict__ fcb, float* __restrict__ mu) {
    __shared__ float sv[SS];
    int blk = blockIdx.x;          // b*35 + j
    int j = blk % NCLS, t = threadIdx.x;
    sv[t] = style[(size_t)blk * SS + t];
    __syncthreads();
    const float* wr = fcw + ((size_t)j * SS + t) * SS;
    float s = fcb[j * SS + t];
    for (int d = 0; d < SS; ++d) s += wr[d] * sv[d];
    mu[(size_t)blk * SS + t] = s > 0.f ? s : 0.f;
}

// ---- K2: per-class conv tables Ag/Ab[b,tap,j,c] = sum_i w[c,i,tap]*mu[b,j,i] --
__global__ void k_tables(const float* __restrict__ cgw, const float* __restrict__ cbw,
                         const float* __restrict__ mu, float* __restrict__ Ag,
                         float* __restrict__ Ab) {
    __shared__ float sMu[SS];
    int blk = blockIdx.x;                  // b*315 + tap*35 + j
    int j = blk % NCLS, tap = (blk / NCLS) % 9, b = blk / (NCLS * 9);
    int t = threadIdx.x;                   // c = 0..255
    if (t < SS) sMu[t] = mu[((size_t)b * NCLS + j) * SS + t];
    __syncthreads();
    float sg = 0.f, sb = 0.f;
    for (int i = 0; i < SS; ++i) {
        float m = sMu[i];
        sg += cgw[((size_t)t * SS + i) * 9 + tap] * m;
        sb += cbw[((size_t)t * SS + i) * 9 + tap] * m;
    }
    size_t o = (size_t)blk * 256 + t;
    Ag[o] = sg; Ab[o] = sb;
}

// ------ K2w: repack sp_gamma_w/sp_beta_w [O,I,3,3] -> bf16 [tap][o][i] --------
__global__ void k_wconv(const float* __restrict__ sgw, const float* __restrict__ sbw,
                        u16* __restrict__ Wg, u16* __restrict__ Wb) {
    int e = blockIdx.x * 256 + threadIdx.x;
    if (e >= 9 * 256 * 128) return;
    int i = e & 127, o = (e >> 7) & 255, tap = e >> 15;
    size_t src = ((size_t)o * 128 + i) * 9 + tap;
    Wg[e] = f2bf(sgw[src]);
    Wb[e] = f2bf(sbw[src]);
}

// ------ K2b: actv[b,h,w,k] = relu(sum_tap mlp_w[k,label(p+tap),tap]+b[k]) -----
__global__ void k_actv(const int* __restrict__ seg, const float* __restrict__ mlpw,
                       const float* __restrict__ mlpb, u16* __restrict__ actv) {
    __shared__ int sJ[18];
    int t = threadIdx.x, blk = blockIdx.x;   // 1024 blocks x 128 pixels
    for (int it = 0; it < 64; ++it) {
        int pix0 = blk * 128 + it * 2;
        if (t < 18) {
            int p = t / 9, tap = t % 9;
            int pix = pix0 + p;
            int b = pix >> 15, rem = pix & 32767, h = rem >> 8, w = rem & 255;
            int hh = h + tap / 3 - 1, ww = w + tap % 3 - 1;
            sJ[t] = (hh >= 0 && hh < SH && ww >= 0 && ww < SW_)
                        ? seg[((size_t)b * SH + hh) * SW_ + ww] : -1;
        }
        __syncthreads();
        int p = t >> 7, k = t & 127;
        float s = mlpb[k];
#pragma unroll
        for (int tap = 0; tap < 9; ++tap) {
            int j = sJ[p * 9 + tap];
            if (j >= 0) s += mlpw[((size_t)k * NCLS + j) * 9 + tap];
        }
        s = s > 0.f ? s : 0.f;
        actv[(size_t)(pix0 + p) * SS + k] = f2bf(s);
        __syncthreads();
    }
}

// ---------------- K3: fused WMMA conv + table gather + IN apply ----------------
// Tile: 64 pixels (one row) x 256 out channels. 512 threads = 16 waves,
// wave = (pxg 0..3)x(chg 0..3): 16px x 64ch -> 4 n-tiles x 2 convs of v8f acc.
// Weight staging is DOUBLE-BUFFERED: next (tap,kc) block's global loads issue
// before the current block's WMMAs, one barrier per K step.
#define AROW 136        // LDS actv pixel stride (halves), conflict-free b128 reads
#define WROW 40         // LDS weight row stride (halves)
#define WBUF 20480      // one weight buffer: 2 convs * 256 * WROW halves (40 KB)
__launch_bounds__(512, 1)
__global__ void k_main(const float* __restrict__ x, const int* __restrict__ seg,
                       const float* __restrict__ Ag, const float* __restrict__ Ab,
                       const float* __restrict__ cgb, const float* __restrict__ cbb,
                       const float* __restrict__ sgb, const float* __restrict__ sbb,
                       const u16* __restrict__ actv, const u16* __restrict__ Wg,
                       const u16* __restrict__ Wb, const float* __restrict__ mean,
                       const float* __restrict__ rstd, const float* __restrict__ pbg,
                       const float* __restrict__ pbb, float* __restrict__ out) {
    __shared__ __align__(16) unsigned char smemU[135776];
    __shared__ int sLbl[3 * 66];
    u16*   sActv = (u16*)smemU;                          // [3][66][AROW] = 53,856 B
    u16*   sW    = (u16*)(smemU + 3 * 66 * AROW * 2);    // 2 x [2][256][WROW] = 81,920 B
    float* sGF   = (float*)smemU;                        // [256][65]
    float* sBF   = (float*)(smemU + 256 * 65 * 4);       // [256][65]

    int tile = blockIdx.x;                // 4 * 128 * 4 = 2048
    int tw = tile & 3, h = (tile >> 2) & 127, b = tile >> 9;
    int wbase = tw * 64;
    int t = threadIdx.x, lane = t & 31, wave = t >> 5;
    int pxg = wave & 3, chg = wave >> 2;
    bool hihalf = lane >= 16;
    int laneN = lane & 15;
    int kA0 = hihalf ? 8 : 0;
    int kB0 = hihalf ? 8 : 0;

    // ---- Phase A: stage labels + actv halo tile (rows h-1..h+1, w-1..w+64) ----
    for (int e = t; e < 198; e += 512) {
        int r = e / 66, p = e % 66;
        int hh = h + r - 1, ww = wbase + p - 1;
        sLbl[e] = (hh >= 0 && hh < SH && ww >= 0 && ww < SW_)
                      ? seg[((size_t)b * SH + hh) * SW_ + ww] : -1;
    }
    for (int e = t; e < 792; e += 512) {     // 198 slots * 4 quarters (64B each)
        int slot = e >> 2, q = e & 3;
        int r = slot / 66, p = slot % 66;
        int hh = h + r - 1, ww = wbase + p - 1;
        uint4 d0 = {0,0,0,0}, d1 = d0, d2 = d0, d3 = d0;
        if (hh >= 0 && hh < SH && ww >= 0 && ww < SW_) {
            const uint4* src = (const uint4*)(actv +
                ((size_t)(((size_t)b * SH + hh) * SW_ + ww)) * SS + q * 32);
            d0 = src[0]; d1 = src[1]; d2 = src[2]; d3 = src[3];
        }
        uint4* dst = (uint4*)(sActv + (size_t)(r * 66 + p) * AROW + q * 32);
        dst[0] = d0; dst[1] = d1; dst[2] = d2; dst[3] = d3;
    }

    // ---- Phase B: implicit-GEMM conv via v_wmma_f32_16x16x32_bf16 ----
    v8f accG[4], accB[4];
#pragma unroll
    for (int n = 0; n < 4; ++n)
#pragma unroll
        for (int i = 0; i < 8; ++i) { accG[n][i] = 0.f; accB[n][i] = 0.f; }

    int pxA = pxg * 16 + laneN;        // A-fragment row (pixel) for this lane
    int convSel = t >> 8, oSel = t & 255;
    const u16* wsrc = (convSel ? Wb : Wg) + (size_t)oSel * 128;
    u16* wdstRow0 = sW + (size_t)(convSel * 256 + oSel) * WROW;

    // prologue: stage step 0 (tap=0,kc=0) into buffer 0
    {
        const uint4* src = (const uint4*)(wsrc);
        uint4* dst = (uint4*)(wdstRow0);
        dst[0] = src[0]; dst[1] = src[1]; dst[2] = src[2]; dst[3] = src[3];
    }
    __syncthreads();

#pragma unroll 2
    for (int it = 0; it < 36; ++it) {
        int tap = it >> 2, kc = it & 3;
        int par = it & 1;
        // issue next step's weight loads early (latency hidden under WMMAs)
        uint4 n0 = {0,0,0,0}, n1 = n0, n2 = n0, n3 = n0;
        bool have = (it + 1) < 36;
        if (have) {
            int it2 = it + 1;
            const uint4* src = (const uint4*)(wsrc +
                ((size_t)(it2 >> 2) * 256) * 128 + (it2 & 3) * 32);
            n0 = src[0]; n1 = src[1]; n2 = src[2]; n3 = src[3];
        }
        // A fragment: lanes 0-15 K={0..7,16..23}, lanes 16-31 K={8..15,24..31}
        int dy = tap / 3, dx = tap % 3;
        const u16* ar = sActv + (size_t)(dy * 66 + pxA + dx) * AROW + kc * 32 + kA0;
        union { uint4 q[2]; v16bf v; } aU;
        aU.q[0] = *(const uint4*)(ar);
        aU.q[1] = *(const uint4*)(ar + 16);
        const u16* sWp = sW + par * WBUF;
#pragma unroll
        for (int nt = 0; nt < 4; ++nt) {
            const u16* wr = sWp + (size_t)(chg * 64 + nt * 16 + laneN) * WROW + kB0;
            union { uint4 q[2]; v16bf v; } bU;
            bU.q[0] = *(const uint4*)(wr);
            bU.q[1] = *(const uint4*)(wr + 16);
            accG[nt] = __builtin_amdgcn_wmma_f32_16x16x32_bf16(
                false, aU.v, false, bU.v, (short)0, accG[nt], false, false);
            const u16* wr2 = wr + 256 * WROW;
            bU.q[0] = *(const uint4*)(wr2);
            bU.q[1] = *(const uint4*)(wr2 + 16);
            accB[nt] = __builtin_amdgcn_wmma_f32_16x16x32_bf16(
                false, aU.v, false, bU.v, (short)0, accB[nt], false, false);
        }
        // store next step's weights into the other buffer
        if (have) {
            uint4* dst = (uint4*)(wdstRow0 + (1 - par) * WBUF);
            dst[0] = n0; dst[1] = n1; dst[2] = n2; dst[3] = n3;
        }
        __syncthreads();   // one barrier per K step (read+write hand-off)
    }

    // ---- Phase C: per-class table gather + blend, transpose via LDS ----
    float ga = 1.f / (1.f + __expf(-pbg[0]));
    float ba = 1.f / (1.f + __expf(-pbb[0]));
    // (last barrier of the K loop already separates sActv/sW reads from aliasing)
#pragma unroll
    for (int nt = 0; nt < 4; ++nt) {
        int c = chg * 64 + nt * 16 + laneN;
        float gbias = cgb[c], bbias = cbb[c];
        float gsbv = sgb[c], bsbv = sbb[c];
#pragma unroll
        for (int v = 0; v < 8; ++v) {
            int pxl = pxg * 16 + v + (hihalf ? 8 : 0);   // C row M -> pixel
            float gav = gbias, bav = bbias;
#pragma unroll
            for (int tap = 0; tap < 9; ++tap) {
                int j = sLbl[(tap / 3) * 66 + pxl + (tap % 3)];
                if (j >= 0) {
                    size_t idx = ((size_t)((b * 9 + tap) * NCLS + j)) * 256 + c;
                    gav += Ag[idx];
                    bav += Ab[idx];
                }
            }
            float gsp = accG[nt][v] + gsbv;
            float bsp = accB[nt][v] + bsbv;
            sGF[c * 65 + pxl] = 1.f + (ga * gav + (1.f - ga) * gsp);
            sBF[c * 65 + pxl] = ba * bav + (1.f - ba) * bsp;
        }
    }
    __syncthreads();

    // ---- Phase D: fused instance-norm apply, coalesced along W ----
    for (int iter = 0; iter < 32; ++iter) {
        int e = iter * 512 + t;
        int c = e >> 6, px = e & 63;
        size_t gi = (((size_t)(b * 256 + c)) * SH + h) * SW_ + wbase + px;
        float xm = mean[b * 256 + c], xr = rstd[b * 256 + c];
        out[gi] = (x[gi] - xm) * xr * sGF[c * 65 + px] + sBF[c * 65 + px];
    }
}

extern "C" void kernel_launch(void* const* d_in, const int* in_sizes, int n_in,
                              void* d_out, int out_size, void* d_ws, size_t ws_size,
                              hipStream_t stream) {
    (void)in_sizes; (void)n_in; (void)out_size; (void)ws_size;
    const float* x     = (const float*)d_in[0];
    const int*   seg   = (const int*)d_in[1];
    const float* style = (const float*)d_in[2];
    const float* fcw   = (const float*)d_in[3];
    const float* fcb   = (const float*)d_in[4];
    const float* cgw   = (const float*)d_in[5];
    const float* cgb   = (const float*)d_in[6];
    const float* cbw   = (const float*)d_in[7];
    const float* cbb   = (const float*)d_in[8];
    const float* mlpw  = (const float*)d_in[9];
    const float* mlpb  = (const float*)d_in[10];
    const float* sgw   = (const float*)d_in[11];
    const float* sgbv  = (const float*)d_in[12];
    const float* sbw   = (const float*)d_in[13];
    const float* sbbv  = (const float*)d_in[14];
    const float* pbg   = (const float*)d_in[15];
    const float* pbb   = (const float*)d_in[16];

    char* ws = (char*)d_ws;
    float* mean = (float*)ws;  ws += 4096;
    float* rstd = (float*)ws;  ws += 4096;
    float* mu   = (float*)ws;  ws += 140 * 128 * 4;          // 71,680
    float* Ag   = (float*)ws;  ws += 4 * 9 * 35 * 256 * 4;   // 1,290,240
    float* Ab   = (float*)ws;  ws += 4 * 9 * 35 * 256 * 4;
    u16*   Wg   = (u16*)ws;    ws += 9 * 256 * 128 * 2;      // 589,824
    u16*   Wb   = (u16*)ws;    ws += 9 * 256 * 128 * 2;
    u16*   actv = (u16*)ws;    ws += (size_t)4 * 128 * 256 * 128 * 2; // 32 MB

    k_stats <<<1024, 256, 0, stream>>>(x, mean, rstd);
    k_mu    <<<140, 128, 0, stream>>>(style, fcw, fcb, mu);
    k_tables<<<1260, 256, 0, stream>>>(cgw, cbw, mu, Ag, Ab);
    k_wconv <<<1152, 256, 0, stream>>>(sgw, sbw, Wg, Wb);
    k_actv  <<<1024, 256, 0, stream>>>(seg, mlpw, mlpb, actv);
    k_main  <<<2048, 512, 0, stream>>>(x, seg, Ag, Ab, cgb, cbb, sgbv, sbbv,
                                       actv, Wg, Wb, mean, rstd, pbg, pbb,
                                       (float*)d_out);
}
